// GCNDecoder_20615843021605
// MI455X (gfx1250) — matrix operand order repaired and checked by reading, not compiled, not run on hardware
//
#include <hip/hip_runtime.h>

// Problem constants (fixed by the reference).
constexpr int Bsz = 16;
constexpr int Esz = 1024;
constexpr int Dsz = 768;
constexpr int Hsz = 768;

typedef __attribute__((ext_vector_type(16))) __bf16 v16bf;
typedef __attribute__((ext_vector_type(8)))  __bf16 v8bf;
typedef __attribute__((ext_vector_type(4)))  __bf16 v4bf;
typedef __attribute__((ext_vector_type(8)))  float  v8f;

// ---------- bf16 helpers (RNE, no libcalls) ----------
static __device__ __forceinline__ __bf16 f2bf(float f) {
    union { float f; unsigned u; } a; a.f = f;
    unsigned r = a.u + 0x7FFFu + ((a.u >> 16) & 1u);
    unsigned short h = (unsigned short)(r >> 16);
    __bf16 o; __builtin_memcpy(&o, &h, 2); return o;
}
static __device__ __forceinline__ float bf2f(__bf16 b) {
    unsigned short h; __builtin_memcpy(&h, &b, 2);
    union { unsigned u; float f; } a; a.u = ((unsigned)h) << 16;
    return a.f;
}
// Fast sigmoid: v_exp_f32 + v_rcp_f32 (avoid precise-division ladder).
static __device__ __forceinline__ float sigmoidf(float x) {
    return __builtin_amdgcn_rcpf(1.0f + __expf(-x));
}

// ---------- WMMA fragment loads ----------
// Byte offsets (u32) off a uniform base -> SADDR+voffset global_load form.
// A (16x32 bf16, M x K): lanes 0-15 row M=lane, K=0..15;
//                        lanes 16-31 row M=lane-16, K=16..31.
static __device__ __forceinline__ v16bf load_a_frag(const char* __restrict__ base,
                                                    unsigned boff) {
    v8bf lo = *(const v8bf*)(base + boff);
    v8bf hi = *(const v8bf*)(base + boff + 16u);
    v16bf a;
#pragma unroll
    for (int i = 0; i < 8; ++i) { a[i] = lo[i]; a[i + 8] = hi[i]; }
    return a;
}
// B (32x16 bf16, K x N), memory holds Brow[n][k] (= B[k][n]):
// lanes 0-15 col N=lane, K=0..15; lanes 16-31 col N=lane-16, K=16..31.
static __device__ __forceinline__ v16bf load_b_frag(const char* __restrict__ base,
                                                    unsigned boff) {
    return *(const v16bf*)(base + boff);
}

// ---------- epilogue ids ----------
// 0: sigmoid*pair + self-loop diag       -> bf16   (A matrix)
// 1: scale rows by dinv[m]               -> bf16   (out = A_norm X)
// 2: + bias[n], relu                     -> bf16   (Hfeat)
// 3: + bias[n]                           -> bf16   (P)
// 4: sigmoid*pair                        -> f32    (Aout, final output)

// Block = 256 threads = 8 waves arranged 4(M) x 2(N); block tile 128x128.
// Wave tile 32(M) x 64(N) = 2x4 WMMA tiles, register double-buffered with a
// peeled tail: the steady-state loop prefetches k+32 before issuing the
// current 8 WMMAs, so VMEM latency hides behind the matrix pipe.
template <int EPI>
__global__ __launch_bounds__(256) void gemm_nt(
    const __bf16* __restrict__ Abase, const __bf16* __restrict__ Bbase,
    int K, int lda, int ldb,
    size_t strideA, size_t strideB,
    void* __restrict__ Cout, int ldc, size_t strideC,
    const int* __restrict__ mask, const float* __restrict__ dinv,
    const float* __restrict__ bias)
{
    const int bz = blockIdx.z;
    const char* A = (const char*)(Abase + (size_t)bz * strideA);
    const char* B = (const char*)(Bbase + (size_t)bz * strideB);

    const int wave = threadIdx.x >> 5;     // 8 waves: 4 (M) x 2 (N)
    const int wm = wave >> 1, wn = wave & 1;
    const int m0 = blockIdx.y * 128 + wm * 32;
    const int n0 = blockIdx.x * 128 + wn * 64;

    // Per-lane base BYTE offsets (all fit in u32).
    const int lane = threadIdx.x & 31;
    const unsigned oA0 = ((unsigned)(m0 + (lane & 15)) * (unsigned)lda
                        + (unsigned)((lane >> 4) << 4)) * 2u;
    const unsigned oA1 = oA0 + 32u * (unsigned)lda;
    const unsigned oB0 = ((unsigned)(n0 + (lane & 15)) * (unsigned)ldb
                        + (unsigned)((lane >> 4) << 4)) * 2u;
    const unsigned bs  = 32u * (unsigned)ldb;      // 16 cols * ldb * 2B

    v8f acc00 = {}, acc01 = {}, acc02 = {}, acc03 = {};
    v8f acc10 = {}, acc11 = {}, acc12 = {}, acc13 = {};

    auto wmma8 = [&](v16bf A0, v16bf A1, v16bf B0, v16bf B1, v16bf B2, v16bf B3) {
        acc00 = __builtin_amdgcn_wmma_f32_16x16x32_bf16(false, A0, false, B0,
                                                        (short)0, acc00, false, false);
        acc10 = __builtin_amdgcn_wmma_f32_16x16x32_bf16(false, A1, false, B0,
                                                        (short)0, acc10, false, false);
        acc01 = __builtin_amdgcn_wmma_f32_16x16x32_bf16(false, A0, false, B1,
                                                        (short)0, acc01, false, false);
        acc11 = __builtin_amdgcn_wmma_f32_16x16x32_bf16(false, A1, false, B1,
                                                        (short)0, acc11, false, false);
        acc02 = __builtin_amdgcn_wmma_f32_16x16x32_bf16(false, A0, false, B2,
                                                        (short)0, acc02, false, false);
        acc12 = __builtin_amdgcn_wmma_f32_16x16x32_bf16(false, A1, false, B2,
                                                        (short)0, acc12, false, false);
        acc03 = __builtin_amdgcn_wmma_f32_16x16x32_bf16(false, A0, false, B3,
                                                        (short)0, acc03, false, false);
        acc13 = __builtin_amdgcn_wmma_f32_16x16x32_bf16(false, A1, false, B3,
                                                        (short)0, acc13, false, false);
    };

    // Prologue: fragments for k=0.
    v16bf a0 = load_a_frag(A, oA0);
    v16bf a1 = load_a_frag(A, oA1);
    v16bf b0 = load_b_frag(B, oB0);
    v16bf b1 = load_b_frag(B, oB0 + bs);
    v16bf b2 = load_b_frag(B, oB0 + 2u * bs);
    v16bf b3 = load_b_frag(B, oB0 + 3u * bs);

    const int nsteps = K >> 5;
    for (int it = 1; it < nsteps; ++it) {
        const unsigned ko = (unsigned)it * 64u;    // 32 k-elements = 64 bytes
        v16bf na0 = load_a_frag(A, oA0 + ko);
        v16bf na1 = load_a_frag(A, oA1 + ko);
        v16bf nb0 = load_b_frag(B, oB0 + ko);
        v16bf nb1 = load_b_frag(B, oB0 + bs + ko);
        v16bf nb2 = load_b_frag(B, oB0 + 2u * bs + ko);
        v16bf nb3 = load_b_frag(B, oB0 + 3u * bs + ko);
        wmma8(a0, a1, b0, b1, b2, b3);
        a0 = na0; a1 = na1; b0 = nb0; b1 = nb1; b2 = nb2; b3 = nb3;
    }
    wmma8(a0, a1, b0, b1, b2, b3);   // peeled tail: no prefetch

    // C layout: VGPR v -> row (v) for lanes 0-15, row (8+v) for lanes 16-31;
    // col = lane & 15.
    const int nn = lane & 15;
    const int mo = (lane >> 4) << 3;

    v8f accs[2][4];
    accs[0][0] = acc00; accs[0][1] = acc01; accs[0][2] = acc02; accs[0][3] = acc03;
    accs[1][0] = acc10; accs[1][1] = acc11; accs[1][2] = acc12; accs[1][3] = acc13;

#pragma unroll
    for (int mi = 0; mi < 2; ++mi) {
#pragma unroll
        for (int ni = 0; ni < 4; ++ni) {
            const int ncol = n0 + ni * 16 + nn;
            float bn = 0.0f, mn = 0.0f;
            if constexpr (EPI == 2 || EPI == 3) bn = bias[ncol];
            if constexpr (EPI == 0 || EPI == 4) mn = (float)mask[(size_t)bz * Esz + ncol];
#pragma unroll
            for (int v = 0; v < 8; ++v) {
                const int mrow = m0 + mi * 16 + mo + v;
                float val = accs[mi][ni][v];
                const size_t cidx = (size_t)bz * strideC + (size_t)mrow * ldc + ncol;
                if constexpr (EPI == 0) {
                    float mm = (float)mask[(size_t)bz * Esz + mrow];
                    val = sigmoidf(val) * mm * mn;
                    if (mrow == ncol) val += mm;           // self-loop on valid nodes
                    ((__bf16*)Cout)[cidx] = f2bf(val);
                } else if constexpr (EPI == 1) {
                    val *= dinv[(size_t)bz * Esz + mrow];
                    ((__bf16*)Cout)[cidx] = f2bf(val);
                } else if constexpr (EPI == 2) {
                    val = fmaxf(val + bn, 0.0f);
                    ((__bf16*)Cout)[cidx] = f2bf(val);
                } else if constexpr (EPI == 3) {
                    ((__bf16*)Cout)[cidx] = f2bf(val + bn);
                } else {
                    float mm = (float)mask[(size_t)bz * Esz + mrow];
                    ((float*)Cout)[cidx] = sigmoidf(val) * mm * mn;
                }
            }
        }
    }
}

// ---------- f32 -> bf16 conversion ----------
__global__ void f32_to_bf16_kernel(const float* __restrict__ in,
                                   __bf16* __restrict__ out, int n) {
    int i = blockIdx.x * blockDim.x + threadIdx.x;
    if (i < n) out[i] = f2bf(in[i]);
}

// ---------- degree rowsum -> dinv = rsqrt(clip(deg)) ----------
__global__ __launch_bounds__(256) void deg_dinv_kernel(const __bf16* __restrict__ Abf,
                                                       float* __restrict__ dinv) {
    __shared__ float red[256];
    const size_t row = blockIdx.x;                  // b*E + e
    const __bf16* p = Abf + row * (size_t)Esz;
    const int t = threadIdx.x;
    v4bf v = *(const v4bf*)(p + t * 4);             // 1024 / 256 = 4 each
    float s = bf2f(v[0]) + bf2f(v[1]) + bf2f(v[2]) + bf2f(v[3]);
    red[t] = s;
    __syncthreads();
    for (int off = 128; off > 0; off >>= 1) {
        if (t < off) red[t] += red[t + off];
        __syncthreads();
    }
    if (t == 0) {
        float deg = fmaxf(red[0], 1e-6f);
        dinv[row] = rsqrtf(deg);
    }
}

// ---------- XsT[b][d][e] = bf16( X[b][e][d] * dinv[b][e] ) ----------
__global__ __launch_bounds__(256) void xst_kernel(const float* __restrict__ X,
                                                  const float* __restrict__ dinv,
                                                  __bf16* __restrict__ XsT) {
    __shared__ float tile[32][33];
    const int b = blockIdx.z;
    const int d0 = blockIdx.x * 32;
    const int e0 = blockIdx.y * 32;
    const int tx = threadIdx.x, ty = threadIdx.y;   // (32, 8)
    const float* Xb = X + (size_t)b * Esz * Dsz;
#pragma unroll
    for (int j = 0; j < 32; j += 8)
        tile[ty + j][tx] = Xb[(size_t)(e0 + ty + j) * Dsz + (d0 + tx)];
    __syncthreads();
    __bf16* out = XsT + (size_t)b * Dsz * Esz;
    const float sc = dinv[b * Esz + e0 + tx];
#pragma unroll
    for (int j = 0; j < 32; j += 8)
        out[(size_t)(d0 + ty + j) * Esz + (e0 + tx)] = f2bf(tile[tx][ty + j] * sc);
}

extern "C" void kernel_launch(void* const* d_in, const int* in_sizes, int n_in,
                              void* d_out, int out_size, void* d_ws, size_t ws_size,
                              hipStream_t stream) {
    (void)in_sizes; (void)n_in; (void)out_size; (void)ws_size;

    const float* X    = (const float*)d_in[0];   // (B,E,D) f32
    const int*   mask = (const int*)  d_in[1];   // (B,E)   i32
    const float* Wg   = (const float*)d_in[2];   // (H,D)
    const float* bg   = (const float*)d_in[3];   // (H,)
    const float* Wp   = (const float*)d_in[4];   // (H,H)
    const float* bp   = (const float*)d_in[5];   // (H,)

    // Workspace layout (bytes). Regions are reused once their producer-consumer
    // chain is done; peak use ~83 MB.
    char* ws = (char*)d_ws;
    const size_t szXbf = (size_t)Bsz * Esz * Dsz * 2;   // 24 MB
    const size_t szAbf = (size_t)Bsz * Esz * Esz * 2;   // 32 MB
    const size_t szXsT = (size_t)Bsz * Dsz * Esz * 2;   // 24 MB
    const size_t szWg  = (size_t)Hsz * Dsz * 2;
    const size_t szWp  = (size_t)Hsz * Hsz * 2;

    __bf16* Xbf  = (__bf16*)(ws);                                   // also: outbf
    __bf16* Abf  = (__bf16*)(ws + szXbf);                           // also: Hfbf
    __bf16* XsT  = (__bf16*)(ws + szXbf + szAbf);                   // also: Pbf
    __bf16* WgB  = (__bf16*)(ws + szXbf + szAbf + szXsT);
    __bf16* WpB  = (__bf16*)(ws + szXbf + szAbf + szXsT + szWg);
    float*  dinv = (float*) (ws + szXbf + szAbf + szXsT + szWg + szWp);

    __bf16* outbf = Xbf;   // A_norm·X  (Xbf dead after GEMM1)
    __bf16* Hfbf  = Abf;   // relu GCN  (Abf dead after GEMM2)
    __bf16* Pbf   = XsT;   // proj      (XsT dead after GEMM2)

    // 1) Convert inputs to bf16.
    {
        int nX = Bsz * Esz * Dsz;
        f32_to_bf16_kernel<<<(nX + 255) / 256, 256, 0, stream>>>(X, Xbf, nX);
        int nW = Hsz * Dsz;
        f32_to_bf16_kernel<<<(nW + 255) / 256, 256, 0, stream>>>(Wg, WgB, nW);
        int nP = Hsz * Hsz;
        f32_to_bf16_kernel<<<(nP + 255) / 256, 256, 0, stream>>>(Wp, WpB, nP);
    }

    // 2) GEMM1: A = sigmoid(X Xᵀ)*pair + diag(m)   (M=N=E, K=D)
    gemm_nt<0><<<dim3(Esz / 128, Esz / 128, Bsz), 256, 0, stream>>>(
        Xbf, Xbf, Dsz, Dsz, Dsz,
        (size_t)Esz * Dsz, (size_t)Esz * Dsz,
        Abf, Esz, (size_t)Esz * Esz, mask, nullptr, nullptr);

    // 3) dinv = rsqrt(clip(rowsum(A), 1e-6))
    deg_dinv_kernel<<<Bsz * Esz, 256, 0, stream>>>(Abf, dinv);

    // 4) XsT[b][d][e] = dinv[b][e] * X[b][e][d]   (bf16, transposed for NT GEMM)
    xst_kernel<<<dim3(Dsz / 32, Esz / 32, Bsz), dim3(32, 8), 0, stream>>>(X, dinv, XsT);

    // 5) GEMM2: out = diag(dinv) · (A · XsTᵀ)     (M=E, N=D, K=E; A symmetric)
    gemm_nt<1><<<dim3(Dsz / 128, Esz / 128, Bsz), 256, 0, stream>>>(
        Abf, XsT, Esz, Esz, Esz,
        (size_t)Esz * Esz, (size_t)Dsz * Esz,
        outbf, Dsz, (size_t)Esz * Dsz, nullptr, dinv, nullptr);

    // 6) GEMM3: Hfeat = relu(out · W_gcnᵀ + b_gcn)  (M=E, N=H, K=D)
    gemm_nt<2><<<dim3(Hsz / 128, Esz / 128, Bsz), 256, 0, stream>>>(
        outbf, WgB, Dsz, Dsz, Dsz,
        (size_t)Esz * Dsz, (size_t)0,
        Hfbf, Hsz, (size_t)Esz * Hsz, nullptr, nullptr, bg);

    // 7) GEMM4: P = Hfeat · W_projᵀ + b_proj        (M=E, N=H, K=H)
    gemm_nt<3><<<dim3(Hsz / 128, Esz / 128, Bsz), 256, 0, stream>>>(
        Hfbf, WpB, Hsz, Hsz, Hsz,
        (size_t)Esz * Hsz, (size_t)0,
        Pbf, Hsz, (size_t)Esz * Hsz, nullptr, nullptr, bp);

    // 8) GEMM5: Aout = sigmoid(P Pᵀ) * pair -> f32 output  (M=N=E, K=H)
    gemm_nt<4><<<dim3(Esz / 128, Esz / 128, Bsz), 256, 0, stream>>>(
        Pbf, Pbf, Hsz, Hsz, Hsz,
        (size_t)Esz * Hsz, (size_t)Esz * Hsz,
        d_out, Esz, (size_t)Esz * Esz, mask, nullptr, nullptr);
}
